// LSTM_baseline_61392262529220
// MI455X (gfx1250) — matrix operand order repaired and checked by reading, not compiled, not run on hardware
//
#include <hip/hip_runtime.h>
#include <hip/hip_bf16.h>

// ---------------------------------------------------------------------------
// LSTM (B=64, L=512, I=256, H=1024) + linear head, CDNA5 / gfx1250 (wave32).
// Single persistent kernel: 32 blocks run all 512 timesteps with a
// device-scope barrier. bf16 WMMA (v_wmma_f32_16x16x32_bf16), f32 accumulate.
// A-panels staged via async global->LDS (ASYNCcnt pipelined: h stage and
// x_{t+1} prefetch overlap the x-part GEMM). Weights stream from L2 (10 MB
// bf16, resident in 192 MB L2 across all steps). c-state lives in registers.
// ---------------------------------------------------------------------------

typedef __attribute__((ext_vector_type(16))) __bf16 v16bf;
typedef __attribute__((ext_vector_type(8)))  float  v8f;

#define BATCH 64
#define SEQL  512
#define IDIM  256
#define HDIM  1024
#define GDIM  4096                 // 4*HDIM (i,f,g,o)
#define XPAD  (IDIM + 8)           // 264 elems: 132 dwords (=4 mod 64 banks)
#define HPAD  (HDIM + 8)           // 1032 elems: 516 dwords (=4 mod 64 banks)
#define NBLK  32                   // persistent blocks (= HDIM/32)

union Frag32 { uint4 u[2]; v16bf v; };   // 32 bytes = 16 bf16

#if defined(__has_builtin)
#if __has_builtin(__builtin_amdgcn_global_load_async_to_lds_b128)
#define HAVE_ASYNC_LDS 1
#endif
#if __has_builtin(__builtin_amdgcn_s_wait_asynccnt)
#define HAVE_WAIT_ASYNC 1
#endif
#endif

// pointer types for the async global->LDS builtin (arg0 AS1 int4*, arg1 AS3)
typedef int v4i_vs __attribute__((vector_size(16)));
typedef __attribute__((address_space(1))) v4i_vs* as1_v4i_p;
typedef __attribute__((address_space(3))) v4i_vs* as3_v4i_p;

__device__ __forceinline__ float sigmoidf_(float x) {
  return 1.0f / (1.0f + __expf(-x));
}

// 16-byte global -> LDS stage (async, ASYNCcnt-tracked)
__device__ __forceinline__ void stage16(const __bf16* __restrict__ g,
                                        __bf16* __restrict__ l) {
#if defined(HAVE_ASYNC_LDS)
  __builtin_amdgcn_global_load_async_to_lds_b128(
      (as1_v4i_p)(void*)const_cast<__bf16*>(g), (as3_v4i_p)(void*)l, 0, 0);
#else
  *(uint4*)l = *(const uint4*)g;
#endif
}

template <int N>
__device__ __forceinline__ void wait_async() {
#if defined(HAVE_WAIT_ASYNC)
  __builtin_amdgcn_s_wait_asynccnt(N);
#else
  asm volatile("s_wait_asynccnt %0" ::"i"(N) : "memory");
#endif
}

// ---------------------------------------------------------------------------
__global__ void k_init(float* __restrict__ accum, __bf16* __restrict__ h0,
                       unsigned* __restrict__ barcnt) {
  int i = blockIdx.x * blockDim.x + threadIdx.x;
  if (i < BATCH * HDIM) h0[i] = (__bf16)0.0f;
  if (i < SEQL * BATCH) accum[i] = 0.0f;
  if (i == 0) *barcnt = 0u;
}

// x (B,L,I) f32 -> xb (L,B,I) bf16
__global__ void k_pack_x(const float* __restrict__ x, __bf16* __restrict__ xb) {
  int i = blockIdx.x * blockDim.x + threadIdx.x;       // over B*L*I
  int ii = i % IDIM;
  int rest = i / IDIM;
  int t = rest % SEQL;
  int b = rest / SEQL;
  xb[((size_t)t * BATCH + b) * IDIM + ii] = (__bf16)x[i];
}

__global__ void k_f32_to_bf16(const float* __restrict__ in,
                              __bf16* __restrict__ out, int n) {
  int i = blockIdx.x * blockDim.x + threadIdx.x;
  if (i < n) out[i] = (__bf16)in[i];
}

// ---------------------------------------------------------------------------
// One WMMA K-step (chunk of 32) over 4 M-tiles against one B fragment.
// ISA bf16 operand layouts (wave32):
//   A 16x32: lane l (m=l&15, hi=l>>4): e 0..7  -> k = 8*hi + e
//                                      e 8..15 -> k = 16 + 8*hi + (e-8)
//   B 32x16: lane l (n=l&15, hi=l>>4): same k mapping as A
//   C 16x16: vgpr r, lane l: m = r + 8*(l>>4), n = l&15
// ---------------------------------------------------------------------------
__device__ __forceinline__ void mma_step_lds(const __bf16* __restrict__ apanel,
                                             int rowstride, int k0,
                                             const __bf16* __restrict__ brow,
                                             int lm, int lhi, v8f cacc[4]) {
  Frag32 bf;
  bf.u[0] = *(const uint4*)(brow + 8 * lhi);
  bf.u[1] = *(const uint4*)(brow + 8 * lhi + 16);
#pragma unroll
  for (int mt = 0; mt < 4; ++mt) {
    const __bf16* aptr = apanel + (size_t)(mt * 16 + lm) * rowstride + k0;
    Frag32 af;
    af.u[0] = *(const uint4*)(aptr + 8 * lhi);        // ds_load_b128
    af.u[1] = *(const uint4*)(aptr + 8 * lhi + 16);   // ds_load_b128
    cacc[mt] = __builtin_amdgcn_wmma_f32_16x16x32_bf16(
        false, af.v, false, bf.v, (short)0, cacc[mt], false, false);
  }
}

// ---------------------------------------------------------------------------
// Persistent LSTM kernel. Grid: 32 blocks x 256 threads (8 waves).
// Block owns h-columns [j0, j0+32); wave w: gate q = w&3, 16-col half w>>2.
// Per timestep:
//   issue async h-stage (32 ops/thr) + x_{t+1} prefetch (8 ops/thr)
//   wait asynccnt<=40 (drains last iter's x) -> x-part GEMM (8 K-chunks)
//   wait asynccnt<=8  (h staged, x prefetch still in flight) -> h-part GEMM
//   LDS gate exchange -> cell update (c in registers) -> h_next + W_lin dot
//   device-scope barrier (leader atomic + acquire spin)
// ---------------------------------------------------------------------------
__global__ __launch_bounds__(256) void k_lstm_persistent(
    const __bf16* __restrict__ xb,    // (L,B,I) bf16
    const __bf16* __restrict__ wih,   // (G,I)   bf16
    const __bf16* __restrict__ whh,   // (G,H)   bf16
    __bf16* __restrict__ h0buf,       // (B,H)   bf16 (even-t source)
    __bf16* __restrict__ h1buf,       // (B,H)   bf16 (odd-t source)
    const float* __restrict__ b_ih, const float* __restrict__ b_hh,
    const float* __restrict__ w_lin,  // (H)
    float* __restrict__ accum,        // (L,B)
    unsigned* __restrict__ barcnt) {
  __shared__ __bf16 xpanel[2][BATCH][XPAD];  // 66 KB  (double-buffered x_t)
  __shared__ __bf16 hpanel[BATCH][HPAD];     // 129 KB (h_prev)
  __shared__ float  gsm[4][BATCH][32];       // 32 KB  gate exchange

  const int tid  = threadIdx.x;
  const int lane = tid & 31;
  const int wv   = tid >> 5;            // wave 0..7
  const int q    = wv & 3;              // gate (i,f,g,o)
  const int nh   = wv >> 2;             // 16-col half of the 32 j-cols
  const int j0   = blockIdx.x * 32;
  const int lm   = lane & 15;
  const int lhi  = lane >> 4;
  const int jc   = nh * 16 + lm;        // column within block (0..31)
  const int grow = q * HDIM + j0 + jc;  // gate row in W (0..4095)

  const __bf16* wihrow = wih + (size_t)grow * IDIM;
  const __bf16* whhrow = whh + (size_t)grow * HDIM;
  const float bias = b_ih[grow] + b_hh[grow];

  // per-thread cell state: b = tid&63, 8 fixed j columns -> registers
  const int bb = tid & 63;
  const int jq = tid >> 6;              // 0..3
  float creg[8];
  float wl[8];
#pragma unroll
  for (int u = 0; u < 8; ++u) {
    creg[u] = 0.0f;
    wl[u] = w_lin[j0 + jq * 8 + u];
  }

  // prologue: prefetch x_0 into xpanel[0]
  for (int c = tid; c < (BATCH * IDIM) / 8; c += 256) {
    int m = c >> 5, col = (c & 31) * 8;
    stage16(xb + c * 8, &xpanel[0][m][col]);
  }

  for (int t = 0; t < SEQL; ++t) {
    const int cur = t & 1, nxt = cur ^ 1;
    const __bf16* hprev = (t & 1) ? h1buf : h0buf;
    __bf16*       hnext = (t & 1) ? h0buf : h1buf;
    const int tn = (t + 1 < SEQL) ? (t + 1) : t;   // keep async counts uniform

    // issue h stage (32 async/thread), then x_{t+1} prefetch (8 async/thread)
    for (int c = tid; c < (BATCH * HDIM) / 8; c += 256) {
      int m = c >> 7, col = (c & 127) * 8;
      stage16(hprev + c * 8, &hpanel[m][col]);
    }
    const __bf16* xsrcn = xb + (size_t)tn * BATCH * IDIM;
    for (int c = tid; c < (BATCH * IDIM) / 8; c += 256) {
      int m = c >> 5, col = (c & 31) * 8;
      stage16(xsrcn + c * 8, &xpanel[nxt][m][col]);
    }
    wait_async<40>();       // drains everything older than this iteration
    __syncthreads();        // xpanel[cur] complete across all waves

    v8f cacc[4] = {};
#pragma unroll
    for (int kk = 0; kk < IDIM / 32; ++kk)
      mma_step_lds(&xpanel[cur][0][0], XPAD, kk * 32, wihrow + kk * 32,
                   lm, lhi, cacc);

    wait_async<8>();        // in-order retire: only the x prefetch remains
    __syncthreads();        // hpanel complete across all waves

#pragma unroll 4
    for (int kk = 0; kk < HDIM / 32; ++kk)
      mma_step_lds(&hpanel[0][0], HPAD, kk * 32, whhrow + kk * 32,
                   lm, lhi, cacc);

    // gate exchange
#pragma unroll
    for (int mt = 0; mt < 4; ++mt) {
#pragma unroll
      for (int r = 0; r < 8; ++r) {
        int m = mt * 16 + 8 * lhi + r;          // batch row (C layout)
        gsm[q][m][jc] = cacc[mt][r] + bias;
      }
    }
    __syncthreads();

    // cell update: c in registers, h to global (bf16), fused W_lin partial
    float partial = 0.0f;
#pragma unroll
    for (int u = 0; u < 8; ++u) {
      int jcc = jq * 8 + u;
      float ig = sigmoidf_(gsm[0][bb][jcc]);
      float fg = sigmoidf_(gsm[1][bb][jcc]);
      float gg = tanhf(gsm[2][bb][jcc]);
      float og = sigmoidf_(gsm[3][bb][jcc]);
      float cnew = fg * creg[u] + ig * gg;
      creg[u] = cnew;
      float hval = og * tanhf(cnew);
      hnext[(size_t)bb * HDIM + j0 + jcc] = (__bf16)hval;
      partial += hval * wl[u];
    }
    atomicAdd(accum + t * BATCH + bb, partial);

    // ---- device-scope barrier ----
    __threadfence();                 // make hnext visible device-wide
    __syncthreads();                 // all waves of block arrived
    if (tid == 0) {
      __hip_atomic_fetch_add(barcnt, 1u, __ATOMIC_ACQ_REL,
                             __HIP_MEMORY_SCOPE_AGENT);
      while (__hip_atomic_load(barcnt, __ATOMIC_ACQUIRE,
                               __HIP_MEMORY_SCOPE_AGENT) <
             (unsigned)NBLK * (unsigned)(t + 1))
        __builtin_amdgcn_s_sleep(1);
    }
    __syncthreads();                 // release whole block past the barrier
  }
}

// out (B,L,1) = tanh(accum + b_lin)
__global__ void k_final(const float* __restrict__ accum,
                        const float* __restrict__ b_lin,
                        float* __restrict__ out) {
  int i = blockIdx.x * blockDim.x + threadIdx.x;    // over B*L
  if (i < BATCH * SEQL) {
    int b = i / SEQL, t = i % SEQL;
    out[i] = tanhf(accum[t * BATCH + b] + b_lin[0]);
  }
}

// ---------------------------------------------------------------------------
extern "C" void kernel_launch(void* const* d_in, const int* in_sizes, int n_in,
                              void* d_out, int out_size, void* d_ws,
                              size_t ws_size, hipStream_t stream) {
  (void)in_sizes; (void)n_in; (void)out_size; (void)ws_size;
  const float* x     = (const float*)d_in[0];
  const float* W_ih  = (const float*)d_in[1];
  const float* W_hh  = (const float*)d_in[2];
  const float* b_ih  = (const float*)d_in[3];
  const float* b_hh  = (const float*)d_in[4];
  const float* W_lin = (const float*)d_in[5];
  const float* b_lin = (const float*)d_in[6];
  float* out = (float*)d_out;

  char* ws = (char*)d_ws;
  size_t off = 0;
  auto alloc = [&](size_t bytes) -> void* {
    off = (off + 255) & ~(size_t)255;
    void* p = ws + off;
    off += bytes;
    return p;
  };
  __bf16*   xb    = (__bf16*)alloc((size_t)SEQL * BATCH * IDIM * 2); // 16.8 MB
  __bf16*   wihb  = (__bf16*)alloc((size_t)GDIM * IDIM * 2);         //  2 MB
  __bf16*   whhb  = (__bf16*)alloc((size_t)GDIM * HDIM * 2);         //  8 MB
  __bf16*   h0    = (__bf16*)alloc((size_t)BATCH * HDIM * 2);
  __bf16*   h1    = (__bf16*)alloc((size_t)BATCH * HDIM * 2);
  float*    accum = (float*)alloc((size_t)SEQL * BATCH * 4);
  unsigned* barc  = (unsigned*)alloc(256);

  k_init<<<(BATCH * HDIM + 255) / 256, 256, 0, stream>>>(accum, h0, barc);
  k_pack_x<<<((size_t)BATCH * SEQL * IDIM + 255) / 256, 256, 0, stream>>>(x, xb);
  k_f32_to_bf16<<<(GDIM * IDIM + 255) / 256, 256, 0, stream>>>(W_ih, wihb,
                                                               GDIM * IDIM);
  k_f32_to_bf16<<<(GDIM * HDIM + 255) / 256, 256, 0, stream>>>(W_hh, whhb,
                                                               GDIM * HDIM);

  k_lstm_persistent<<<NBLK, 256, 0, stream>>>(xb, wihb, whhb, h0, h1,
                                              b_ih, b_hh, W_lin, accum, barc);

  k_final<<<(BATCH * SEQL + 255) / 256, 256, 0, stream>>>(accum, b_lin, out);
}
